// ComplexAttention_79886391705966
// MI455X (gfx1250) — matrix operand order repaired and checked by reading, not compiled, not run on hardware
//
#include <hip/hip_runtime.h>
#include <hip/hip_bf16.h>

// Problem dims (fixed by reference): B=4, T=1024, S=32, D=512, 2D=1024
#define BT_   4096
#define S_    32
#define D_    512
#define TWO_D 1024

typedef __attribute__((ext_vector_type(16))) __bf16 v16bf;
typedef __attribute__((ext_vector_type(8)))  float  v8f;

union Frag16 { v16bf v; uint4 q[2]; };

// A-fragment (16-bit A 16x32 ISA layout):
//   lanes 0-15 (khalf=0): K = k0+{0..7} and k0+{16..23}
//   lanes 16-31 (khalf=1): K = k0+{8..15} and k0+{24..31}
// rowbase points at element [row][k0] of a row-major bf16 buffer.
__device__ inline v16bf frag_a(const __bf16* rowbase, int khalf) {
  Frag16 f;
  const uint4* p = (const uint4*)rowbase;   // 16B = 8 bf16 per uint4
  f.q[0] = p[khalf];
  f.q[1] = p[2 + khalf];
  return f.v;
}

// B-fragment (16-bit B 32x16): lanes 0-15 hold K=k0+{0..7} and k0+{16..23}, lanes 16-31 K=k0+{8..15} and k0+{24..31},
// read from the N-major (transposed) weight buffer (same K order as frag_a).
__device__ inline v16bf frag_b(const __bf16* rowbase, int khalf) {
  Frag16 f;
  const uint4* p = (const uint4*)rowbase;
  f.q[0] = p[khalf];
  f.q[1] = p[2 + khalf];
  return f.v;
}

// ---------------- prep kernels -------------------------------------------

// Wt[n][k] = (bf16) W[k][n];  N is a power of two, nshift = log2(N)
__global__ void k_transpose_bf16(const float* __restrict__ W,
                                 __bf16* __restrict__ Wt,
                                 int K, int N, int nshift) {
  long idx = (long)blockIdx.x * 256 + threadIdx.x;
  if (idx >= (long)K * N) return;
  int k = (int)(idx >> nshift);
  int n = (int)(idx & (N - 1));
  Wt[(long)n * K + k] = (__bf16)W[idx];
}

// cur2bf[row][c] = bf16( c<512 ? real[row][c] : imag[row][c-512] )
__global__ void k_concat_bf16(const float* __restrict__ re,
                              const float* __restrict__ im,
                              __bf16* __restrict__ out) {
  long idx = (long)blockIdx.x * 256 + threadIdx.x;   // BT_*TWO_D total
  int row = (int)(idx >> 10);
  int c   = (int)(idx & 1023);
  float v = (c < D_) ? re[(long)row * D_ + c] : im[(long)row * D_ + (c - D_)];
  out[idx] = (__bf16)v;
}

// ---------------- Q projection GEMM (WMMA) --------------------------------
// Q[4096,512] = cur2bf[4096,1024] @ qW ; one 16x16 tile per wave.
__global__ __launch_bounds__(256)
void k_qproj(const __bf16* __restrict__ A,      // [4096][1024] bf16
             const __bf16* __restrict__ Bt,     // qWt [512][1024] bf16 (N-major)
             const float*  __restrict__ qb,
             float* __restrict__ Q) {
  int lane = threadIdx.x & 31;
  int wave = threadIdx.x >> 5;
  int tile = blockIdx.x * 8 + wave;              // 8192 tiles = 256 mt x 32 nt
  int mt = tile >> 5;
  int nt = tile & 31;
  int khalf = lane >> 4;
  int lr    = lane & 15;

  const __bf16* arow = A  + (long)(mt * 16 + lr) * TWO_D;
  const __bf16* brow = Bt + (long)(nt * 16 + lr) * TWO_D;

  v8f acc = {};
#pragma unroll 4
  for (int k0 = 0; k0 < TWO_D; k0 += 32) {
    v16bf a = frag_a(arow + k0, khalf);
    v16bf b = frag_b(brow + k0, khalf);
    acc = __builtin_amdgcn_wmma_f32_16x16x32_bf16(false, a, false, b,
                                                  (short)0, acc, false, false);
  }
  int col  = nt * 16 + lr;
  int row0 = mt * 16 + khalf * 8;        // C layout: lanes0-15 M=0..7, lanes16-31 M=8..15
  float bias = qb[col];
#pragma unroll
  for (int r = 0; r < 8; ++r)
    Q[(long)(row0 + r) * D_ + col] = acc[r] + bias;
}

// ---------------- fused attention kernel ----------------------------------
// One block per (b,t). 256 threads = 8 waves.
// LDS: Hs (32 x 1032 bf16), Ks (32 x 512 bf16), Qs (512 f32),
//      part (32x8 f32), wgt (32 f32), ctx (1024 f32)  => 106112 bytes.
#define HROW 1032
#define KROW 512
#define OFF_H    0
#define OFF_K    66048
#define OFF_Q    98816
#define OFF_PART 100864
#define OFF_WGT  101888
#define OFF_CTX  102016
#define SMEM_BYTES 106112

__global__ __launch_bounds__(256)
void k_attn(const float* __restrict__ hist_real,
            const float* __restrict__ hist_imag,
            const float* __restrict__ cur_real,
            const float* __restrict__ cur_imag,
            const float* __restrict__ confidence,
            const __bf16* __restrict__ kWt,   // [512][1024]  bf16 N-major
            const float*  __restrict__ kb,
            const __bf16* __restrict__ vWt,   // [1024][1024] bf16 N-major
            const float*  __restrict__ vb,
            const float*  __restrict__ Qg,    // [4096][512] f32
            float* __restrict__ out) {        // interleaved complex64 [4096][512]
  extern __shared__ char smem[];
  __bf16* Hs   = (__bf16*)(smem + OFF_H);
  __bf16* Ks   = (__bf16*)(smem + OFF_K);
  float*  Qs   = (float*)(smem + OFF_Q);
  float*  part = (float*)(smem + OFF_PART);
  float*  wgt  = (float*)(smem + OFF_WGT);
  float*  ctxL = (float*)(smem + OFF_CTX);

  int tid  = threadIdx.x;
  int bid  = blockIdx.x;                 // b*T + t
  int lane = tid & 31;
  int wave = tid >> 5;
  int khalf = lane >> 4;
  int lr    = lane & 15;

  // ---- Phase 0: stage H (f32 -> bf16) into LDS, stage Q (f32) ------------
  const float* hr = hist_real + (long)bid * S_ * D_;
  const float* hi = hist_imag + (long)bid * S_ * D_;
#pragma unroll 4
  for (int it = 0; it < 32; ++it) {
    int e = (it * 256 + tid) * 4;        // 32768 bf16 elements total
    int s = e >> 10;
    int c = e & 1023;
    const float* src = (c < D_) ? (hr + s * D_ + c) : (hi + s * D_ + (c - D_));
    float4 v = *(const float4*)src;
    union { __bf16 h[4]; uint2 u; } pk;
    pk.h[0] = (__bf16)v.x; pk.h[1] = (__bf16)v.y;
    pk.h[2] = (__bf16)v.z; pk.h[3] = (__bf16)v.w;
    *(uint2*)&Hs[s * HROW + c] = pk.u;
  }
  Qs[tid]       = Qg[(long)bid * D_ + tid];
  Qs[tid + 256] = Qg[(long)bid * D_ + tid + 256];
  __syncthreads();

  const __bf16* a0 = Hs + lr * HROW;          // M-tile 0, this lane's row
  const __bf16* a1 = Hs + (16 + lr) * HROW;   // M-tile 1

  // ---- Phase A: K projection via WMMA, K -> LDS (bf16) -------------------
  for (int i = 0; i < 4; ++i) {
    int nt   = wave + i * 8;                  // 0..31, uniform per wave
    int ncol = nt * 16;
    const __bf16* brow = kWt + (long)(ncol + lr) * TWO_D;
    v8f acc0 = {}, acc1 = {};
#pragma unroll 4
    for (int k0 = 0; k0 < TWO_D; k0 += 32) {
      __builtin_prefetch(brow + k0 + 512, 0, 1);     // global_prefetch_b8
      v16bf b  = frag_b(brow + k0, khalf);
      v16bf m0 = frag_a(a0 + k0, khalf);
      v16bf m1 = frag_a(a1 + k0, khalf);
      acc0 = __builtin_amdgcn_wmma_f32_16x16x32_bf16(false, m0, false, b,
                                                     (short)0, acc0, false, false);
      acc1 = __builtin_amdgcn_wmma_f32_16x16x32_bf16(false, m1, false, b,
                                                     (short)0, acc1, false, false);
    }
    int col = ncol + lr;
    float bs = kb[col];
    int r0 = khalf * 8;
#pragma unroll
    for (int r = 0; r < 8; ++r) {
      Ks[(r0 + r) * KROW + col]      = (__bf16)(acc0[r] + bs);  // s = 0..15
      Ks[(16 + r0 + r) * KROW + col] = (__bf16)(acc1[r] + bs);  // s = 16..31
    }
  }
  __syncthreads();

  // ---- Phase B: scores = (Q . K) * scale * conf ; stable softmax ---------
  {
    int s   = tid >> 3;                 // 8 threads per history slot
    int sub = tid & 7;
    const __bf16* krow = Ks + s * KROW;
    float p = 0.f;
#pragma unroll 8
    for (int j = 0; j < 64; ++j) {
      int e = sub * 64 + j;
      p += Qs[e] * (float)krow[e];
    }
    part[s * 8 + sub] = p;
  }
  __syncthreads();
  if (tid < 32) {                        // wave 0, all 32 lanes
    float sc = 0.f;
#pragma unroll
    for (int u = 0; u < 8; ++u) sc += part[tid * 8 + u];
    float conf = confidence[bid];
    sc *= 0.044194173824159216f * conf;  // 512^-0.5
    float mx = sc;
    for (int m = 16; m >= 1; m >>= 1) mx = fmaxf(mx, __shfl_xor(mx, m, 32));
    float ex = __expf(sc - mx);
    float sm = ex;
    for (int m = 16; m >= 1; m >>= 1) sm += __shfl_xor(sm, m, 32);
    wgt[tid] = ex / sm;
  }
  __syncthreads();

  float w_lo = wgt[khalf * 8 + 0];       // touch to keep wgt resident (optional)
  (void)w_lo;

  // ---- Phase C: V projection fused with ctx = sum_s w[s] * V[s,:] --------
  // Each wave owns whole V column tiles -> reduce over s entirely in regs.
  // sum_s w[s] = 1, so vb folds in exactly: ctx = vb + sum_s w[s]*Vraw.
  for (int i = 0; i < 8; ++i) {
    int nt   = wave + i * 8;             // 0..63, uniform per wave
    int ncol = nt * 16;
    const __bf16* brow = vWt + (long)(ncol + lr) * TWO_D;
    v8f acc0 = {}, acc1 = {};
#pragma unroll 4
    for (int k0 = 0; k0 < TWO_D; k0 += 32) {
      __builtin_prefetch(brow + k0 + 512, 0, 1);
      v16bf b  = frag_b(brow + k0, khalf);
      v16bf m0 = frag_a(a0 + k0, khalf);
      v16bf m1 = frag_a(a1 + k0, khalf);
      acc0 = __builtin_amdgcn_wmma_f32_16x16x32_bf16(false, m0, false, b,
                                                     (short)0, acc0, false, false);
      acc1 = __builtin_amdgcn_wmma_f32_16x16x32_bf16(false, m1, false, b,
                                                     (short)0, acc1, false, false);
    }
    int r0 = khalf * 8;
    float c0 = 0.f;
#pragma unroll
    for (int r = 0; r < 8; ++r) {
      c0 += wgt[r0 + r]      * acc0[r];  // s = r0+r
      c0 += wgt[16 + r0 + r] * acc1[r];  // s = 16+r0+r
    }
    c0 += __shfl_xor(c0, 16, 32);        // fold khalf pair (same column)
    if (lane < 16) {
      int col = ncol + lr;
      ctxL[col] = vb[col] + c0;
    }
  }
  __syncthreads();

  // ---- Phase D: out = current + 0.1*ctx, interleaved complex64 -----------
  long obase = (long)bid * D_;
  float2* ov = (float2*)out;
  {
    float r0v = cur_real[obase + tid] + 0.1f * ctxL[tid];
    float i0v = cur_imag[obase + tid] + 0.1f * ctxL[tid + 512];
    ov[obase + tid] = make_float2(r0v, i0v);
    float r1v = cur_real[obase + tid + 256] + 0.1f * ctxL[tid + 256];
    float i1v = cur_imag[obase + tid + 256] + 0.1f * ctxL[tid + 768];
    ov[obase + tid + 256] = make_float2(r1v, i1v);
  }
}

// ---------------- launch ---------------------------------------------------
extern "C" void kernel_launch(void* const* d_in, const int* in_sizes, int n_in,
                              void* d_out, int out_size, void* d_ws, size_t ws_size,
                              hipStream_t stream) {
  const float* hist_real = (const float*)d_in[0];
  const float* hist_imag = (const float*)d_in[1];
  const float* cur_real  = (const float*)d_in[2];
  const float* cur_imag  = (const float*)d_in[3];
  const float* conf      = (const float*)d_in[4];
  const float* qW        = (const float*)d_in[5];
  const float* qb        = (const float*)d_in[6];
  const float* kW        = (const float*)d_in[7];
  const float* kb        = (const float*)d_in[8];
  const float* vW        = (const float*)d_in[9];
  const float* vb        = (const float*)d_in[10];

  // Workspace layout (needs >= 20,971,520 bytes):
  //   [0,          8388608)  Q   f32 [4096][512]
  //   [8388608,    9437184)  kWt bf16 [512][1024]
  //   [9437184,   11534336)  vWt bf16 [1024][1024]
  //   [11534336,  12582912)  qWt bf16 [512][1024]
  //   [12582912,  20971520)  cur2 bf16 [4096][1024]
  char* ws = (char*)d_ws;
  float*  Qbuf   = (float*)ws;
  __bf16* kWt    = (__bf16*)(ws + 8388608);
  __bf16* vWt    = (__bf16*)(ws + 9437184);
  __bf16* qWt    = (__bf16*)(ws + 11534336);
  __bf16* cur2bf = (__bf16*)(ws + 12582912);

  // Prep: bf16 + [N,K] transpose of weights, bf16 concat of current.
  k_transpose_bf16<<<2048, 256, 0, stream>>>(kW, kWt, TWO_D, D_, 9);
  k_transpose_bf16<<<4096, 256, 0, stream>>>(vW, vWt, TWO_D, TWO_D, 10);
  k_transpose_bf16<<<2048, 256, 0, stream>>>(qW, qWt, TWO_D, D_, 9);
  k_concat_bf16<<<16384, 256, 0, stream>>>(cur_real, cur_imag, cur2bf);

  // Q = cur2 @ qW + qb  (WMMA GEMM)
  k_qproj<<<1024, 256, 0, stream>>>(cur2bf, qWt, qb, Qbuf);

  // Fused K/V projection + softmax attention, one block per (b,t).
  k_attn<<<BT_, 256, SMEM_BYTES, stream>>>(hist_real, hist_imag,
                                           cur_real, cur_imag, conf,
                                           kWt, kb, vWt, vb, Qbuf,
                                           (float*)d_out);
}
